// QuantumWordMatrix_22935125360975
// MI455X (gfx1250) — compile-verified, matrix operand order checked by reading, not run on hardware
//
#include <hip/hip_runtime.h>
#include <hip/hip_bf16.h>
#include <stdint.h>

#define THRESH 0.33f
#define TEPS 1e-10f

typedef int v8i __attribute__((ext_vector_type(8)));

__device__ __forceinline__ int tern(float x, float thr) {
    return (x > thr) - (x < -thr);
}
__device__ __forceinline__ unsigned pack4(int b0, int b1, int b2, int b3) {
    return (unsigned)(b0 & 255) | ((unsigned)(b1 & 255) << 8) |
           ((unsigned)(b2 & 255) << 16) | ((unsigned)(b3 & 255) << 24);
}

// ---------------- Kernel 0: ternarize query -> sign plane and (3|b|-2) plane --
__global__ void qwm_prep(const float* __restrict__ q, int D,
                         signed char* __restrict__ qsign,
                         signed char* __restrict__ qbias,
                         float* __restrict__ hdr) {
    __shared__ float redf[256];
    __shared__ int   redi[256];
    int t = threadIdx.x;
    float m = 0.f;
    for (int i = t; i < D; i += 256) m = fmaxf(m, fabsf(q[i]));
    redf[t] = m; __syncthreads();
    for (int s = 128; s; s >>= 1) { if (t < s) redf[t] = fmaxf(redf[t], redf[t + s]); __syncthreads(); }
    float thr = THRESH * (redf[0] + TEPS);
    __syncthreads();
    int cnt = 0;
    for (int i = t; i < D; i += 256) {
        float x = q[i];
        int s = tern(x, thr);
        int a = s < 0 ? -s : s;
        qsign[i] = (signed char)s;
        qbias[i] = (signed char)(3 * a - 2);   // folds -2*sum|a| into the WMMA
        cnt += a;
    }
    redi[t] = cnt; __syncthreads();
    for (int s = 128; s; s >>= 1) { if (t < s) redi[t] += redi[t + s]; __syncthreads(); }
    if (t == 0) hdr[0] = (float)redi[0];   // sum |b|
}

// ---------------- Kernel 1: coarse ternary scores via V_WMMA_I32_16X16X64_IU8 -
// 256 threads = 8 waves; each wave scores a 16-row tile (WG covers 128 rows).
// S' = dot(a,b) + dot(|a|, 3|b|-2) = P + 3Q - 2*sum|a|
// sim = 0.75*S' + D - 1.5*sum|b|
__global__ void qwm_coarse(const float* __restrict__ W, int V, int D,
                           const signed char* __restrict__ qsign,  // qbias = qsign + D
                           const float* __restrict__ hdr,
                           float* __restrict__ sims) {
    __shared__ __align__(16) signed char qlds[4096];   // [0,D): sign  [D,2D): bias
    const int lane = threadIdx.x & 31;
    const int wave = threadIdx.x >> 5;

    // ---- Stage the 2*D-byte query planes into LDS via async copy (wave 0) ---
    if (wave == 0) {
        for (int off = lane * 16; off < 2 * D; off += 512) {
            unsigned loff = (unsigned)(uintptr_t)(&qlds[off]);   // low 32b = LDS offset
            const signed char* src = qsign + off;
            asm volatile("global_load_async_to_lds_b128 %0, %1, off"
                         :: "v"(loff), "v"(src) : "memory");
        }
        asm volatile("s_wait_asynccnt 0x0" ::: "memory");
    }
    __syncthreads();

    const int rowBase = (blockIdx.x * 8 + wave) * 16;
    if (rowBase + 16 > V) return;                 // wave-uniform: EXEC all-1s below
    const float simBias = (float)D - 1.5f * hdr[0];
    const float* tile = W + (size_t)rowBase * D;

    // ---- Pass A: per-row max|x|, fully coalesced float4 streaming -----------
    const int myRow = lane & 15;                  // row this lane owns in the A-fragment
    const int hh = lane >> 4;                     // lane half (K-set selector)
    float myRowMax = 0.f;
    const float4* t4 = (const float4*)tile;
    const int itPerRow = D / 128;                 // float4s per row / 32 lanes
    for (int r = 0; r < 16; ++r) {
        float mm = 0.f;
        for (int i = 0; i < itPerRow; ++i) {
            float4 v = t4[(size_t)(r * itPerRow + i) * 32 + lane];
            mm = fmaxf(mm, fmaxf(fmaxf(fabsf(v.x), fabsf(v.y)),
                                 fmaxf(fabsf(v.z), fabsf(v.w))));
        }
        for (int off = 16; off; off >>= 1) mm = fmaxf(mm, __shfl_xor(mm, off, 32));
        if (r == myRow) myRowMax = mm;
    }
    const float thr = THRESH * (myRowMax + TEPS);

    // ---- Prefetch this lane's pass-B row into the WGP cache (L2 -> L0) ------
    const float* rowp = tile + (size_t)myRow * D;
    for (int kk = 0; kk < D; kk += 64) __builtin_prefetch(rowp + kk, 0, 3);

    // ---- Pass B: ternarize (L2/WGP$-resident re-read) + dual-plane IU8 WMMA -
    v8i acc = {};
#pragma unroll 2
    for (int kk = 0; kk < D; kk += 64) {
        v8i As, Aa, Bs, Ba;
#pragma unroll
        for (int i = 0; i < 8; ++i) {
            // ISA 8-bit A 16x64 K mapping: vgpr i, lane-half hh -> 4 consecutive K
            const int kl = ((i >> 1) << 4) + ((i & 1) << 2) + (hh << 3);
            float4 x = *(const float4*)(rowp + kk + kl);
            int s0 = tern(x.x, thr), s1 = tern(x.y, thr);
            int s2 = tern(x.z, thr), s3 = tern(x.w, thr);
            int a0 = s0 < 0 ? -s0 : s0, a1 = s1 < 0 ? -s1 : s1;
            int a2 = s2 < 0 ? -s2 : s2, a3 = s3 < 0 ? -s3 : s3;
            As[i] = (int)pack4(s0, s1, s2, s3);
            Aa[i] = (int)pack4(a0, a1, a2, a3);
            Bs[i] = *(const int*)(qlds + kk + kl);       // 16 identical query cols (ds_load)
            Ba[i] = *(const int*)(qlds + D + kk + kl);
        }
        acc = __builtin_amdgcn_wmma_i32_16x16x64_iu8(true, As, true, Bs, acc, false, false);
        acc = __builtin_amdgcn_wmma_i32_16x16x64_iu8(true, Aa, true, Ba, acc, false, false);
    }

    // ---- N=0 column lives wholly in lanes 0 (rows 0-7) and 16 (rows 8-15) ---
    if (lane == 0) {
        float4 f0, f1;
        f0.x = 0.75f * (float)acc[0] + simBias;  f0.y = 0.75f * (float)acc[1] + simBias;
        f0.z = 0.75f * (float)acc[2] + simBias;  f0.w = 0.75f * (float)acc[3] + simBias;
        f1.x = 0.75f * (float)acc[4] + simBias;  f1.y = 0.75f * (float)acc[5] + simBias;
        f1.z = 0.75f * (float)acc[6] + simBias;  f1.w = 0.75f * (float)acc[7] + simBias;
        *(float4*)(sims + rowBase)     = f0;
        *(float4*)(sims + rowBase + 4) = f1;
    }
    if (lane == 16) {
        float4 f0, f1;
        f0.x = 0.75f * (float)acc[0] + simBias;  f0.y = 0.75f * (float)acc[1] + simBias;
        f0.z = 0.75f * (float)acc[2] + simBias;  f0.w = 0.75f * (float)acc[3] + simBias;
        f1.x = 0.75f * (float)acc[4] + simBias;  f1.y = 0.75f * (float)acc[5] + simBias;
        f1.z = 0.75f * (float)acc[6] + simBias;  f1.w = 0.75f * (float)acc[7] + simBias;
        *(float4*)(sims + rowBase + 8)  = f0;
        *(float4*)(sims + rowBase + 12) = f1;
    }
}

// ---------------- Kernel 2: top-3k coarse select, cosine refine, top-k -------
__global__ void qwm_select(const float* __restrict__ q,
                           const float* __restrict__ W, int V, int D,
                           const int* __restrict__ topk_p,
                           float* __restrict__ sims,
                           float* __restrict__ out) {
    const int t = threadIdx.x;
    const int NT = 1024;
    __shared__ float rv[1024];
    __shared__ int   ri[1024];
    __shared__ int   candIdx[192];
    __shared__ float refined[192];
    __shared__ float dotS;
    __shared__ float qn;

    int tk = *topk_p; if (tk > 64) tk = 64; if (tk < 1) tk = 1;
    int k3 = tk * 3;  if (k3 > V) k3 = V;  if (k3 > 192) k3 = 192;

    // ||q||
    float ss = 0.f;
    for (int i = t; i < D; i += NT) { float x = q[i]; ss += x * x; }
    rv[t] = ss; __syncthreads();
    for (int s = 512; s; s >>= 1) { if (t < s) rv[t] += rv[t + s]; __syncthreads(); }
    if (t == 0) qn = sqrtf(rv[0]) + 1e-12f;
    __syncthreads();

    // k3 masked argmax sweeps over sims (resident in L2)
    for (int j = 0; j < k3; ++j) {
        float bm = -3.4e38f; int bi = 0;
        for (int i = t; i < V; i += NT) {
            float v = sims[i];
            if (v > bm) { bm = v; bi = i; }
        }
        rv[t] = bm; ri[t] = bi; __syncthreads();
        for (int s = 512; s; s >>= 1) {
            if (t < s) {
                if (rv[t + s] > rv[t] || (rv[t + s] == rv[t] && ri[t + s] < ri[t])) {
                    rv[t] = rv[t + s]; ri[t] = ri[t + s];
                }
            }
            __syncthreads();
        }
        if (t == 0) { candIdx[j] = ri[0]; sims[ri[0]] = -3.4e38f; }
        __syncthreads();
    }

    // cosine refine on the k3 candidates
    for (int j = 0; j < k3; ++j) {
        const float* row = W + (size_t)candIdx[j] * D;
        float d = 0.f, s2 = 0.f;
        for (int i = t; i < D; i += NT) { float x = row[i]; d += x * q[i]; s2 += x * x; }
        rv[t] = d; __syncthreads();
        for (int s = 512; s; s >>= 1) { if (t < s) rv[t] += rv[t + s]; __syncthreads(); }
        if (t == 0) dotS = rv[0];
        __syncthreads();
        rv[t] = s2; __syncthreads();
        for (int s = 512; s; s >>= 1) { if (t < s) rv[t] += rv[t + s]; __syncthreads(); }
        if (t == 0) refined[j] = dotS / ((sqrtf(rv[0]) + 1e-12f) * qn);
        __syncthreads();
    }

    // final top-k over k3 (serial: tiny)
    if (t == 0) {
        for (int j = 0; j < tk; ++j) {
            int best = 0; float bv = -3.4e38f;
            for (int i = 0; i < k3; ++i)
                if (refined[i] > bv) { bv = refined[i]; best = i; }
            out[j]      = bv;                       // scores
            out[tk + j] = (float)candIdx[best];     // global indices
            refined[best] = -3.4e38f;
        }
    }
}

extern "C" void kernel_launch(void* const* d_in, const int* in_sizes, int n_in,
                              void* d_out, int out_size, void* d_ws, size_t ws_size,
                              hipStream_t stream) {
    const float* q  = (const float*)d_in[0];
    const float* W  = (const float*)d_in[1];
    const int*   tk = (const int*)d_in[2];
    const int D = in_sizes[0];
    const int V = (int)((long long)in_sizes[1] / D);

    char* ws = (char*)d_ws;
    float*       sims  = (float*)ws;                       // V floats
    signed char* qsign = (signed char*)(ws + (size_t)V * 4);
    signed char* qbias = qsign + D;                        // must stay adjacent
    float*       hdr   = (float*)(qbias + D);

    qwm_prep<<<1, 256, 0, stream>>>(q, D, qsign, qbias, hdr);

    const int rowsPerBlock = 128;                          // 8 waves x 16 rows
    const int nb = (V + rowsPerBlock - 1) / rowsPerBlock;
    qwm_coarse<<<nb, 256, 0, stream>>>(W, V, D, qsign, hdr, sims);

    qwm_select<<<1, 1024, 0, stream>>>(q, W, V, D, tk, sims, (float*)d_out);
}
